// GT32dim_2_modes_3_layers_91113436217385
// MI455X (gfx1250) — compile-verified, matrix operand order checked by reading, not run on hardware
//
#include <hip/hip_runtime.h>
#include <hip/hip_bf16.h>

// ---------------------------------------------------------------------------
// CDNA5 (gfx1250) wave32 WMMA types
// ---------------------------------------------------------------------------
typedef __attribute__((ext_vector_type(16))) _Float16 v16h;
typedef __attribute__((ext_vector_type(8)))  float    v8f;

#define C32 32
#define NCLS 5

// order-preserving float<->uint map for atomicMax on floats (incl. negatives)
__device__ __forceinline__ unsigned fmap(float f) {
  unsigned u = __float_as_uint(f);
  return (u & 0x80000000u) ? ~u : (u | 0x80000000u);
}
__device__ __forceinline__ float funmap(unsigned u) {
  unsigned v = (u & 0x80000000u) ? (u & 0x7FFFFFFFu) : ~u;
  return __uint_as_float(v);
}

// ---------------------------------------------------------------------------
// WMMA operand builders (ISA 7.12.2 layouts, wave32)
// A (16xK f16 tile): lane half (lane>>4) picks K runs {h*8..h*8+7, 16+h*8..}
// B (Kx16 f16 tile): lane holds column n, same K-halfword ordering as A
// A rows are 32B-aligned here -> use b128 vector loads.
// ---------------------------------------------------------------------------
__device__ __forceinline__ v16h loadA32(const float* __restrict__ X, int row,
                                        int stride, int coff, int half) {
  const float4* __restrict__ p =
      (const float4*)(X + (size_t)row * stride + coff + half * 8);
  const float4 a0 = p[0], a1 = p[1];   // K run 0: 8 floats
  const float4 a2 = p[4], a3 = p[5];   // K run 1: +16 floats
  v16h a;
  a[0] = (_Float16)a0.x;  a[1] = (_Float16)a0.y;
  a[2] = (_Float16)a0.z;  a[3] = (_Float16)a0.w;
  a[4] = (_Float16)a1.x;  a[5] = (_Float16)a1.y;
  a[6] = (_Float16)a1.z;  a[7] = (_Float16)a1.w;
  a[8] = (_Float16)a2.x;  a[9] = (_Float16)a2.y;
  a[10] = (_Float16)a2.z; a[11] = (_Float16)a2.w;
  a[12] = (_Float16)a3.x; a[13] = (_Float16)a3.y;
  a[14] = (_Float16)a3.z; a[15] = (_Float16)a3.w;
  return a;
}

__device__ __forceinline__ v16h loadB32(const float* __restrict__ W, int ldb,
                                        int n, int half, int koff) {
  v16h b;
#pragma unroll
  for (int j = 0; j < 8; ++j)
    b[j] = (_Float16)W[(size_t)(koff + half * 8 + j) * ldb + n];
#pragma unroll
  for (int j = 0; j < 8; ++j)
    b[8 + j] = (_Float16)W[(size_t)(koff + 16 + half * 8 + j) * ldb + n];
  return b;
}

// ---------------------------------------------------------------------------
// One WMMA (x2 column halves) + bias + store; wave-uniform full-tile fast path
// ---------------------------------------------------------------------------
__device__ __forceinline__ void wmma_store32(const v16h& A,
                                             const float* __restrict__ W,
                                             const float* __restrict__ bias,
                                             float* __restrict__ out,
                                             int tileBase, int lane, int nNodes,
                                             bool full) {
  const int half = lane >> 4, nn = lane & 15;
#pragma unroll
  for (int ch = 0; ch < 2; ++ch) {
    v16h B = loadB32(W, C32, ch * 16 + nn, half, 0);
    v8f c = {};
    c = __builtin_amdgcn_wmma_f32_16x16x32_f16(false, A, false, B, (short)0, c,
                                               false, false);
    const float bb = bias[ch * 16 + nn];
    float* __restrict__ o =
        out + (size_t)(tileBase + half * 8) * C32 + ch * 16 + nn;
    if (full) {                       // wave-uniform branch
#pragma unroll
      for (int r = 0; r < 8; ++r) o[(size_t)r * C32] = c[r] + bb;
    } else {
#pragma unroll
      for (int r = 0; r < 8; ++r)
        if (tileBase + half * 8 + r < nNodes) o[(size_t)r * C32] = c[r] + bb;
    }
  }
}

// ---------------------------------------------------------------------------
// Per-layer node projections: qn = Xd@Wq+bq, kn = Xs@Wk+bk, vn = Xs@Wv+bv,
// skip(out init) = Xd@Ws+bs.   One wave = 16 nodes, 8 WMMAs.
// ---------------------------------------------------------------------------
__global__ __launch_bounds__(256) void k_node_gemm(
    const float* __restrict__ srcX, int sStride, int sOff,
    const float* __restrict__ dstX, int dStride, int dOff,
    const float* __restrict__ Wq, const float* __restrict__ bq,
    const float* __restrict__ Wk, const float* __restrict__ bk,
    const float* __restrict__ Wv, const float* __restrict__ bv,
    const float* __restrict__ Ws, const float* __restrict__ bs,
    float* __restrict__ qn, float* __restrict__ kn, float* __restrict__ vn,
    float* __restrict__ skip, int nNodes) {
  const int wave = (int)((blockIdx.x * blockDim.x + threadIdx.x) >> 5);
  const int lane = threadIdx.x & 31;
  const int ntiles = (nNodes + 15) >> 4;
  if (wave >= ntiles) return;  // wave-uniform: EXEC stays all-ones for WMMA
  const int half = lane >> 4;
  const int tileBase = wave * 16;
  const bool full = (tileBase + 16 <= nNodes);
  int row = tileBase + (lane & 15);
  if (row > nNodes - 1) row = nNodes - 1;  // clamp loads only (no EXEC change)

  const v16h Ad = loadA32(dstX, row, dStride, dOff, half);
  const v16h As = loadA32(srcX, row, sStride, sOff, half);

  wmma_store32(Ad, Wq, bq, qn, tileBase, lane, nNodes, full);
  wmma_store32(As, Wk, bk, kn, tileBase, lane, nNodes, full);
  wmma_store32(As, Wv, bv, vn, tileBase, lane, nNodes, full);
  wmma_store32(Ad, Ws, bs, skip, tileBase, lane, nNodes, full);  // agg init
}

// ---------------------------------------------------------------------------
// eproj collapse: csum[e][c] = sum_k We[e][k][c]
// ---------------------------------------------------------------------------
__global__ void k_csum(const float* __restrict__ We, float* __restrict__ csum) {
  const int t = blockIdx.x * blockDim.x + threadIdx.x;
  if (t >= 3 * C32) return;
  const int e = t >> 5, c = t & 31;
  float s = 0.f;
#pragma unroll
  for (int k = 0; k < C32; ++k) s += We[(size_t)e * 1024 + k * C32 + c];
  csum[t] = s;
}

__global__ void k_init_attn(unsigned* __restrict__ amax, float* __restrict__ den,
                            int n) {
  const int t = blockIdx.x * blockDim.x + threadIdx.x;
  if (t < n) { amax[t] = 0x007FFFFFu; /* fmap(-inf) */ den[t] = 0.f; }
}

// pass A: a_e = q.(k+eproj) / sqrt(C), segment max over dst
__global__ __launch_bounds__(256) void k_score(
    const float* __restrict__ qn, const float* __restrict__ kn,
    const int* __restrict__ src, const int* __restrict__ dst,
    const float* __restrict__ eattr, const float* __restrict__ csum_l,
    const float* __restrict__ be_l, float* __restrict__ aout,
    unsigned* __restrict__ amax, int E) {
  const int e = blockIdx.x * blockDim.x + threadIdx.x;
  if (e >= E) return;
  const int s = src[e], d = dst[e];
  const float4* __restrict__ q4 = (const float4*)(qn + (size_t)d * C32);
  const float4* __restrict__ k4 = (const float4*)(kn + (size_t)s * C32);
  float a;
  if (eattr) {
    const float4* __restrict__ cs4 = (const float4*)csum_l;
    const float4* __restrict__ be4 = (const float4*)be_l;
    float qk = 0.f, qcs = 0.f, qbe = 0.f;
#pragma unroll
    for (int c = 0; c < 8; ++c) {
      const float4 qv = q4[c], kv = k4[c], cv = cs4[c], bv = be4[c];
      qk  += qv.x * kv.x + qv.y * kv.y + qv.z * kv.z + qv.w * kv.w;
      qcs += qv.x * cv.x + qv.y * cv.y + qv.z * cv.z + qv.w * cv.w;
      qbe += qv.x * bv.x + qv.y * bv.y + qv.z * bv.z + qv.w * bv.w;
    }
    a = qk + eattr[e] * qcs + qbe;
  } else {
    float qk = 0.f;
#pragma unroll
    for (int c = 0; c < 8; ++c) {
      const float4 qv = q4[c], kv = k4[c];
      qk += qv.x * kv.x + qv.y * kv.y + qv.z * kv.z + qv.w * kv.w;
    }
    a = qk;
  }
  a *= 0.17677669529663687f;  // 1/sqrt(32)
  aout[e] = a;
  atomicMax(&amax[d], fmap(a));
}

// pass B: ex = exp(a - amax[dst]); den[dst] += ex
__global__ __launch_bounds__(256) void k_exp(
    float* __restrict__ aex, const int* __restrict__ dst,
    const unsigned* __restrict__ amax, float* __restrict__ den, int E) {
  const int e = blockIdx.x * blockDim.x + threadIdx.x;
  if (e >= E) return;
  const int d = dst[e];
  const float ex = expf(aex[e] - funmap(amax[d]));
  aex[e] = ex;
  atomicAdd(&den[d], ex);
}

// pass C: out[dst] += (v[src] + eproj) * ex/den[dst]   (out pre-init = skip)
__global__ __launch_bounds__(256) void k_agg(
    const float* __restrict__ vn, const float* __restrict__ aex,
    const float* __restrict__ den, const int* __restrict__ src,
    const int* __restrict__ dst, const float* __restrict__ eattr,
    const float* __restrict__ csum_l, const float* __restrict__ be_l,
    float* __restrict__ out, int E) {
  const int e = blockIdx.x * blockDim.x + threadIdx.x;
  if (e >= E) return;
  const int s = src[e], d = dst[e];
  const float w = aex[e] / den[d];
  const float4* __restrict__ v4 = (const float4*)(vn + (size_t)s * C32);
  float* __restrict__ o = out + (size_t)d * C32;
  if (eattr) {
    const float ea = eattr[e];
    const float4* __restrict__ cs4 = (const float4*)csum_l;
    const float4* __restrict__ be4 = (const float4*)be_l;
#pragma unroll
    for (int c = 0; c < 8; ++c) {
      const float4 vv = v4[c], cv = cs4[c], bv = be4[c];
      atomicAdd(&o[c * 4 + 0], (vv.x + ea * cv.x + bv.x) * w);
      atomicAdd(&o[c * 4 + 1], (vv.y + ea * cv.y + bv.y) * w);
      atomicAdd(&o[c * 4 + 2], (vv.z + ea * cv.z + bv.z) * w);
      atomicAdd(&o[c * 4 + 3], (vv.w + ea * cv.w + bv.w) * w);
    }
  } else {
#pragma unroll
    for (int c = 0; c < 8; ++c) {
      const float4 vv = v4[c];
      atomicAdd(&o[c * 4 + 0], vv.x * w);
      atomicAdd(&o[c * 4 + 1], vv.y * w);
      atomicAdd(&o[c * 4 + 2], vv.z * w);
      atomicAdd(&o[c * 4 + 3], vv.w * w);
    }
  }
}

__global__ void k_tanh(float* __restrict__ x, size_t n) {
  const size_t i = (size_t)blockIdx.x * blockDim.x + threadIdx.x;
  if (i < n) x[i] = tanhf(x[i]);
}

__global__ void k_zero(float* __restrict__ p, size_t n) {
  const size_t i = (size_t)blockIdx.x * blockDim.x + threadIdx.x;
  if (i < n) p[i] = 0.f;
}

// segment-sum pooling of the 6 concatenated states into [G,192]
__global__ __launch_bounds__(256) void k_pool(
    const float* __restrict__ states, const int* __restrict__ batch,
    float* __restrict__ pooled, float* __restrict__ cnt, int N) {
  const int nid = blockIdx.x * blockDim.x + threadIdx.x;
  if (nid >= N) return;
  const int b = batch[nid];
  atomicAdd(&cnt[b], 1.f);
  const size_t NC = (size_t)N * C32;
  float* __restrict__ pr = pooled + (size_t)b * 192;
#pragma unroll
  for (int l = 0; l < 6; ++l) {
    const float4* __restrict__ st4 =
        (const float4*)(states + (size_t)l * NC + (size_t)nid * C32);
#pragma unroll
    for (int c = 0; c < 8; ++c) {
      const float4 sv = st4[c];
      atomicAdd(&pr[l * C32 + c * 4 + 0], sv.x);
      atomicAdd(&pr[l * C32 + c * 4 + 1], sv.y);
      atomicAdd(&pr[l * C32 + c * 4 + 2], sv.z);
      atomicAdd(&pr[l * C32 + c * 4 + 3], sv.w);
    }
  }
}

__global__ void k_mean(const float* __restrict__ pooled,
                       const float* __restrict__ cnt,
                       float* __restrict__ mean192, int G) {
  const int t = blockIdx.x * blockDim.x + threadIdx.x;
  if (t >= G * 192) return;
  float c = cnt[t / 192];
  c = (c > 0.f) ? c : 1.f;
  mean192[t] = pooled[t] / c;
}

// head W1 via WMMA: [G,192] @ [192,96]   (pool commutes with affine map)
__global__ __launch_bounds__(256) void k_head_w1(
    const float* __restrict__ mean192, const float* __restrict__ cnt,
    const float* __restrict__ W1, const float* __restrict__ b1,
    float* __restrict__ h1, int G) {
  const int wave = (int)((blockIdx.x * blockDim.x + threadIdx.x) >> 5);
  const int lane = threadIdx.x & 31;
  const int ntiles = G >> 4;
  if (wave >= ntiles) return;  // wave-uniform
  const int half = lane >> 4, nn = lane & 15;
  const int tileBase = wave * 16;
  const int row = tileBase + (lane & 15);

  v16h A[6];
#pragma unroll
  for (int kb = 0; kb < 6; ++kb) A[kb] = loadA32(mean192, row, 192, kb * 32, half);

#pragma unroll
  for (int nt = 0; nt < 6; ++nt) {
    const int n = nt * 16 + nn;
    v8f c = {};
#pragma unroll
    for (int kb = 0; kb < 6; ++kb) {
      v16h B = loadB32(W1, 96, n, half, kb * 32);
      c = __builtin_amdgcn_wmma_f32_16x16x32_f16(false, A[kb], false, B,
                                                 (short)0, c, false, false);
    }
#pragma unroll
    for (int r = 0; r < 8; ++r) {
      const int rrow = tileBase + half * 8 + r;
      const float bias = (cnt[rrow] > 0.f) ? b1[n] : 0.f;
      h1[(size_t)rrow * 96 + n] = c[r] + bias;
    }
  }
}

// tiny MLP tail + log_softmax, one block per graph
__global__ __launch_bounds__(96) void k_head_tail(
    const float* __restrict__ h1, const float* __restrict__ W2,
    const float* __restrict__ b2, const float* __restrict__ W3,
    const float* __restrict__ b3, const float* __restrict__ W4,
    const float* __restrict__ b4, const float* __restrict__ W5,
    const float* __restrict__ b5, const float* __restrict__ W6,
    const float* __restrict__ b6, float* __restrict__ out) {
  __shared__ float t1[96], t2[48], t3[48], t4[24], t5[24], lg[NCLS];
  const int g = blockIdx.x, t = threadIdx.x;
  if (t < 96) t1[t] = h1[(size_t)g * 96 + t];
  __syncthreads();
  if (t < 48) {
    float s = b2[t];
    for (int k = 0; k < 96; ++k) s += t1[k] * W2[k * 48 + t];
    t2[t] = fmaxf(s, 0.f);
  }
  __syncthreads();
  if (t < 48) {
    float s = b3[t];
    for (int k = 0; k < 48; ++k) s += t2[k] * W3[k * 48 + t];
    t3[t] = s;
  }
  __syncthreads();
  if (t < 24) {
    float s = b4[t];
    for (int k = 0; k < 48; ++k) s += t3[k] * W4[k * 24 + t];
    t4[t] = s;
  }
  __syncthreads();
  if (t < 24) {
    float s = b5[t];
    for (int k = 0; k < 24; ++k) s += t4[k] * W5[k * 24 + t];
    t5[t] = s;
  }
  __syncthreads();
  if (t < NCLS) {
    float s = b6[t];
    for (int k = 0; k < 24; ++k) s += t5[k] * W6[k * NCLS + t];
    lg[t] = s;
  }
  __syncthreads();
  if (t == 0) {
    float mx = lg[0];
    for (int j = 1; j < NCLS; ++j) mx = fmaxf(mx, lg[j]);
    float se = 0.f;
    for (int j = 0; j < NCLS; ++j) se += expf(lg[j] - mx);
    const float lse = mx + logf(se);
    for (int j = 0; j < NCLS; ++j) out[(size_t)g * NCLS + j] = lg[j] - lse;
  }
}

// ---------------------------------------------------------------------------
// host launcher
// ---------------------------------------------------------------------------
extern "C" void kernel_launch(void* const* d_in, const int* in_sizes, int n_in,
                              void* d_out, int out_size, void* d_ws,
                              size_t ws_size, hipStream_t stream) {
  const float* inter_emb = (const float*)d_in[0];
  const int*   ii        = (const int*)d_in[1];
  const float* eattr     = (const float*)d_in[2];
  const int*   ui        = (const int*)d_in[3];
  const float* uv_emb    = (const float*)d_in[4];
  const int*   batch     = (const int*)d_in[5];
  const float* Wq = (const float*)d_in[6],  *bq = (const float*)d_in[7];
  const float* Wk = (const float*)d_in[8],  *bk = (const float*)d_in[9];
  const float* Wv = (const float*)d_in[10], *bv = (const float*)d_in[11];
  const float* Ws = (const float*)d_in[12], *bs = (const float*)d_in[13];
  const float* We = (const float*)d_in[14], *be = (const float*)d_in[15];
  const float* W1 = (const float*)d_in[16], *b1 = (const float*)d_in[17];
  const float* W2 = (const float*)d_in[18], *b2 = (const float*)d_in[19];
  const float* W3 = (const float*)d_in[20], *b3 = (const float*)d_in[21];
  const float* W4 = (const float*)d_in[22], *b4 = (const float*)d_in[23];
  const float* W5 = (const float*)d_in[24], *b5 = (const float*)d_in[25];
  const float* W6 = (const float*)d_in[26], *b6 = (const float*)d_in[27];
  float* out = (float*)d_out;

  const int N = in_sizes[4] / C32;   // uv_target_emb rows
  const int E = in_sizes[2];         // edges per set
  const int G = out_size / NCLS;     // graphs
  const size_t NC = (size_t)N * C32;

  // workspace carve-up (floats)
  float* states  = (float*)d_ws;            // 6*N*32
  float* qn      = states + 6 * NC;         // N*32
  float* kn      = qn + NC;                 // N*32
  float* vn      = kn + NC;                 // N*32
  float* aex     = vn + NC;                 // E
  unsigned* amax = (unsigned*)(aex + E);    // N
  float* den     = (float*)(amax + N);      // N
  float* csum    = den + N;                 // 3*32
  float* pooled  = csum + 96;               // G*192
  float* cnt     = pooled + (size_t)G * 192;// G
  float* mean192 = cnt + G;                 // G*192
  float* h1      = mean192 + (size_t)G * 192;// G*96

  const int TB = 256;
  const int gN   = (N + TB - 1) / TB;
  const int gE   = (E + TB - 1) / TB;
  const int gNC  = (int)((NC + TB - 1) / TB);
  const int gemmBlocks = ((N + 15) / 16 * 32 + TB - 1) / TB;   // 1 wave / 16 nodes
  const int w1Blocks   = (G / 16 * 32 + TB - 1) / TB;

  k_csum<<<1, 96, 0, stream>>>(We, csum);

  for (int i = 0; i < 6; ++i) {
    const bool even = (i % 2 == 0);
    // layer feature sources
    const float* srcX; int sStride, sOff;
    const float* dstX; int dStride, dOff;
    if (i == 0) {
      srcX = inter_emb; sStride = 64; sOff = 0;    // f0
      dstX = inter_emb; dStride = 64; dOff = 32;   // f1
    } else if (even) {
      srcX = states + (size_t)(i - 1) * NC; sStride = 32; sOff = 0;
      dstX = srcX;                          dStride = 32; dOff = 0;
    } else {
      srcX = states + (size_t)(i - 1) * NC; sStride = 32; sOff = 0;
      dstX = uv_emb;                        dStride = 32; dOff = 0;
    }
    float* agg = states + (size_t)i * NC;  // skip-init, then scatter, then tanh

    k_node_gemm<<<gemmBlocks, TB, 0, stream>>>(
        srcX, sStride, sOff, dstX, dStride, dOff,
        Wq + (size_t)i * 1024, bq + i * 32, Wk + (size_t)i * 1024, bk + i * 32,
        Wv + (size_t)i * 1024, bv + i * 32, Ws + (size_t)i * 1024, bs + i * 32,
        qn, kn, vn, agg, N);

    const int* src = even ? ii : ui;
    const int* dst = even ? (ii + E) : (ui + E);
    const float* ea    = even ? eattr : nullptr;
    const float* cs_l  = even ? (csum + (i / 2) * 32) : nullptr;
    const float* be_l  = even ? (be + (i / 2) * 32) : nullptr;

    k_init_attn<<<gN, TB, 0, stream>>>(amax, den, N);
    k_score<<<gE, TB, 0, stream>>>(qn, kn, src, dst, ea, cs_l, be_l, aex, amax, E);
    k_exp<<<gE, TB, 0, stream>>>(aex, dst, amax, den, E);
    k_agg<<<gE, TB, 0, stream>>>(vn, aex, den, src, dst, ea, cs_l, be_l, agg, E);
    k_tanh<<<gNC, TB, 0, stream>>>(agg, NC);
  }

  const size_t poolN = (size_t)G * 192 + G;  // pooled + cnt (contiguous)
  k_zero<<<(int)((poolN + TB - 1) / TB), TB, 0, stream>>>(pooled, poolN);
  k_pool<<<gN, TB, 0, stream>>>(states, batch, pooled, cnt, N);
  k_mean<<<(G * 192 + TB - 1) / TB, TB, 0, stream>>>(pooled, cnt, mean192, G);
  k_head_w1<<<w1Blocks, TB, 0, stream>>>(mean192, cnt, W1, b1, h1, G);
  k_head_tail<<<G, 96, 0, stream>>>(h1, W2, b2, W3, b3, W4, b4, W5, b5, W6, b6,
                                    out);
}
